// DRMM_class_12524124635901
// MI455X (gfx1250) — matrix operand order repaired and checked by reading, not compiled, not run on hardware
//
#include <hip/hip_runtime.h>

typedef __attribute__((ext_vector_type(16))) _Float16 v16h;
typedef __attribute__((ext_vector_type(8)))  _Float16 v8h;
typedef __attribute__((ext_vector_type(8)))  float    v8f;

#define B      64
#define TQ     32
#define TD     4096
#define H      300
#define HP     320      // H padded to a multiple of the WMMA K=32
#define KT     (HP / 32)
#define NBINS  29
#define NODES  5
#define HSTRIDE 32      // padded per-(b,q) histogram row (30 slots used)
#define TD_PER_BLOCK 512
#define TILES_PER_WAVE 8

// ---------------------------------------------------------------------------
// Kernel 1: L2-normalize each embedding row (cosine prep) and convert to f16,
// padding H=300 -> HP=320 with zeros. One wave32 per vocab row.
// ---------------------------------------------------------------------------
__global__ void __launch_bounds__(256)
k_norm_emb(const float* __restrict__ emb, _Float16* __restrict__ embn, int V) {
  const int row  = (int)(blockIdx.x * (blockDim.x >> 5) + (threadIdx.x >> 5));
  const int lane = threadIdx.x & 31;
  if (row >= V) return;
  const float* src = emb + (size_t)row * H;
  float x[KT];
  float ss = 0.f;
#pragma unroll
  for (int i = 0; i < KT; ++i) {
    const int h = lane + i * 32;
    x[i] = (h < H) ? src[h] : 0.f;
    ss += x[i] * x[i];
  }
#pragma unroll
  for (int off = 16; off > 0; off >>= 1) ss += __shfl_xor(ss, off, 32);
  const float inv = 1.f / (sqrtf(ss + 1e-7f) + 1e-7f);   // matches reference norm
  _Float16* dst = embn + (size_t)row * HP;
#pragma unroll
  for (int i = 0; i < KT; ++i)
    dst[lane + i * 32] = (_Float16)(x[i] * inv);
}

// ---------------------------------------------------------------------------
// Kernel 2: cosine-sim tiles via v_wmma_f32_16x16x32_f16 + LDS histogram.
// Block = batch b x 512 doc cols, 8 waves. Each wave has a FIXED mtile
// (wv & 1), so the whole A operand (10 k-tiles, 80 VGPRs) is loaded once and
// stays register-resident. Per doc tile we preload ALL 10 B fragments, with a
// sched_barrier fencing the load clause from the WMMA chain so LLVM cannot
// re-serialize them into load->wait(0)->wmma pairs; the 20 b128 loads overlap
// in flight and the 10 WMMAs issue back-to-back. Doc tokens for the wave's 8
// tiles are prefetched up front to kill the per-tile b32->wait(0) stall.
// Fragment layouts used here (contiguous k order in A and B, not the ISA table order):
//   A 16x32 f16 : lane L -> row m=L&15; halves k0+(L>>4)*16+{0..7} (elts 0-7)
//                 and k0+8+(L>>4)*16+{0..7} (elts 8-15)  -> two 16B loads
//   B 32x16 f16 : lane L -> col n=L&15; halves k0+(L>>4)*16+{0..15} -> one 32B load
//   C/D 16x16 f32: vgpr r -> m = r + (L>>4)*8, n = L&15
// ---------------------------------------------------------------------------
__global__ void __launch_bounds__(256)
k_sim_hist(const int* __restrict__ sentence, const int* __restrict__ qsent,
           const _Float16* __restrict__ embn, unsigned* __restrict__ counts) {
  __shared__ unsigned lhist[TQ * HSTRIDE];
  const int tid  = threadIdx.x;
  const int b    = blockIdx.x;
  const int td0  = blockIdx.y * TD_PER_BLOCK;
  const int wv   = tid >> 5;
  const int lane = tid & 31;
  const int l15  = lane & 15;
  const int hi   = lane >> 4;

  for (int i = tid; i < TQ * HSTRIDE; i += 256) lhist[i] = 0u;
  __syncthreads();

  // Wave tile assignment: mtile fixed per wave; 8 ntiles, stride 4.
  const int mtile = wv & 1;
  const int mrow  = mtile * 16 + l15;                 // query row this lane feeds
  const int qtok  = qsent[b * TQ + mrow];
  const _Float16* qrow = embn + (size_t)qtok * HP;

  // A operand: resident for the whole kernel (10 k-tiles).
  union AV { v16h v; v8h h[2]; };
  AV afr[KT];
#pragma unroll
  for (int kt = 0; kt < KT; ++kt) {
    afr[kt].h[0] = *(const v8h*)(qrow + kt * 32 + hi * 16);
    afr[kt].h[1] = *(const v8h*)(qrow + kt * 32 + 8 + hi * 16);
  }

  // Prefetch this wave's doc tokens (one per tile; lane -> column l15).
  int dtoks[TILES_PER_WAVE];
#pragma unroll
  for (int i = 0; i < TILES_PER_WAVE; ++i) {
    const int ntile = (wv >> 1) + i * 4;
    dtoks[i] = sentence[b * TD + td0 + ntile * 16 + l15];
  }

  for (int i = 0; i < TILES_PER_WAVE; ++i) {
    const int dtok = dtoks[i];
    const _Float16* drow = embn + (size_t)dtok * HP;

    // Preload the full B operand for this tile (10 x 32B per lane), then
    // fence: no WMMA may be scheduled before all loads are issued.
    v16h bf[KT];
#pragma unroll
    for (int kt = 0; kt < KT; ++kt)
      bf[kt] = *(const v16h*)(drow + kt * 32 + hi * 16);
    __builtin_amdgcn_sched_barrier(0);

    v8f acc = {};
#pragma unroll
    for (int kt = 0; kt < KT; ++kt)
      acc = __builtin_amdgcn_wmma_f32_16x16x32_f16(
          false, afr[kt].v, false, bf[kt], (short)0, acc, false, false);

    // Histogram: bin = searchsorted(linspace(-1,1,30)[1:], sim, 'right')
    //          = floor((sim+1)*29/2); bin 29 == out-of-range (mode='drop').
    // Masked doc positions (token 0) get +1e7 in the reference -> dropped: skip.
    if (dtok != 0) {
#pragma unroll
      for (int r = 0; r < 8; ++r) {
        const int   m = mtile * 16 + r + hi * 8;
        const float s = acc[r];
        int idx = (int)floorf((s + 1.f) * 14.5f);
        if (idx < 0) idx = 0;
        if (idx <= NBINS - 1) atomicAdd(&lhist[m * HSTRIDE + idx], 1u);
        if (s > 0.999f && s < 1.001f) atomicAdd(&lhist[m * HSTRIDE + NBINS], 1u);
      }
    }
  }

  __syncthreads();
  for (int i = tid; i < TQ * (NBINS + 1); i += 256) {
    const int q = i / (NBINS + 1);
    const int s = i % (NBINS + 1);
    const unsigned v = lhist[q * HSTRIDE + s];
    if (v) atomicAdd(&counts[(size_t)(b * TQ + q) * HSTRIDE + s], v);
  }
}

// ---------------------------------------------------------------------------
// Kernel 3: log-hist -> 30x5x1 tanh MLP -> IDF-gated softmax -> output.
// One wave32 per batch, lane == query term (exactly Tq=32 lanes).
// ---------------------------------------------------------------------------
__global__ void __launch_bounds__(256)
k_finalize(const unsigned* __restrict__ counts, const int* __restrict__ qsent,
           const float* __restrict__ idf,
           const float* __restrict__ w1, const float* __restrict__ b1,
           const float* __restrict__ w2, const float* __restrict__ b2,
           const float* __restrict__ gate_w,
           const float* __restrict__ out_w, const float* __restrict__ out_b,
           float* __restrict__ out) {
  const int b = (int)(blockIdx.x * (blockDim.x >> 5) + (threadIdx.x >> 5));
  const int q = threadIdx.x & 31;
  if (b >= B) return;

  const unsigned* crow = counts + (size_t)(b * TQ + q) * HSTRIDE;
  float hist[NBINS + 1];
#pragma unroll
  for (int i = 0; i < NBINS + 1; ++i) hist[i] = logf((float)crow[i] + 1.f);

  float t1[NODES];
#pragma unroll
  for (int j = 0; j < NODES; ++j) {
    float s = b1[j];
    for (int i = 0; i < NBINS + 1; ++i) s += hist[i] * w1[i * NODES + j];
    t1[j] = tanhf(s);
  }
  float f = b2[0];
#pragma unroll
  for (int j = 0; j < NODES; ++j) f += t1[j] * w2[j];
  const float ffw = tanhf(f);

  const int qt = qsent[b * TQ + q];
  float gl = idf[b * TQ + q] * gate_w[0] + (qt == 0 ? -1e7f : 0.f);

  float mx = gl;
#pragma unroll
  for (int off = 16; off > 0; off >>= 1) mx = fmaxf(mx, __shfl_xor(mx, off, 32));
  float e = expf(gl - mx);
  float se = e;
#pragma unroll
  for (int off = 16; off > 0; off >>= 1) se += __shfl_xor(se, off, 32);
  float val = (e / se) * ffw;
#pragma unroll
  for (int off = 16; off > 0; off >>= 1) val += __shfl_xor(val, off, 32);

  if (q == 0) out[b] = val * out_w[0] + out_b[0];
}

// ---------------------------------------------------------------------------
extern "C" void kernel_launch(void* const* d_in, const int* in_sizes, int n_in,
                              void* d_out, int out_size, void* d_ws, size_t ws_size,
                              hipStream_t stream) {
  (void)n_in; (void)out_size; (void)ws_size;
  const int*   sentence = (const int*)  d_in[0];
  const int*   qsent    = (const int*)  d_in[1];
  const float* idf      = (const float*)d_in[2];
  const float* emb      = (const float*)d_in[3];
  const float* w1       = (const float*)d_in[4];
  const float* b1       = (const float*)d_in[5];
  const float* w2       = (const float*)d_in[6];
  const float* b2       = (const float*)d_in[7];
  const float* gate_w   = (const float*)d_in[8];
  const float* out_w    = (const float*)d_in[9];
  const float* out_b    = (const float*)d_in[10];
  float* out = (float*)d_out;

  const int V = in_sizes[3] / H;   // 100000

  // Workspace layout:
  //   [0, V*HP*2)            : normalized f16 embedding table (~64 MB, fits L2)
  //   [V*HP*2, +B*TQ*32*4)   : per-(b,q) histogram counts (u32)
  _Float16* embn   = (_Float16*)d_ws;
  unsigned* counts = (unsigned*)((char*)d_ws + (size_t)V * HP * sizeof(_Float16));

  hipMemsetAsync(counts, 0, (size_t)B * TQ * HSTRIDE * sizeof(unsigned), stream);
  k_norm_emb<<<(V + 7) / 8, 256, 0, stream>>>(emb, embn, V);
  k_sim_hist<<<dim3(B, TD / TD_PER_BLOCK), 256, 0, stream>>>(sentence, qsent, embn, counts);
  k_finalize<<<B / 8, 256, 0, stream>>>(counts, qsent, idf, w1, b1, w2, b2,
                                        gate_w, out_w, out_b, out);
}
